// Decoder_1108101563238
// MI455X (gfx1250) — matrix operand order repaired and checked by reading, not compiled, not run on hardware
//
#include <hip/hip_runtime.h>
#include <hip/hip_bf16.h>

// Dimensions from the reference
#define BB 256      // batch
#define TT 128      // time
#define FF 32       // input features
#define UE 512      // encoder width
#define DD 512      // LSTM width
#define AA 512      // attention width

// gfx1250 async Global->LDS DMA (ASYNCcnt-tracked), guarded so the file still
// compiles if the toolchain lacks the builtins.
#if defined(__HIP_DEVICE_COMPILE__) && \
    __has_builtin(__builtin_amdgcn_global_load_async_to_lds_b128) && \
    __has_builtin(__builtin_amdgcn_s_wait_asynccnt)
#define HAS_ASYNC_LDS 1
#else
#define HAS_ASYNC_LDS 0
#endif

#if HAS_ASYNC_LDS
// Matches clang's diagnosed parameter type: pointer to 16-byte int vector.
typedef __attribute__((__vector_size__(4 * sizeof(int)))) int async_v4i;
typedef __attribute__((address_space(3))) async_v4i* lds_v4i_ptr;
#endif

typedef __attribute__((ext_vector_type(16))) __bf16 v16bf;
typedef __attribute__((ext_vector_type(8)))  float  v8f;
typedef __attribute__((ext_vector_type(4)))  float  vf4;
typedef __attribute__((ext_vector_type(4)))  unsigned int vu4;

__device__ __forceinline__ unsigned int f2bf(float f) {
    unsigned int u = __float_as_uint(f);
    unsigned int r = u + 0x7FFFu + ((u >> 16) & 1u);   // round-to-nearest-even
    return r >> 16;
}
__device__ __forceinline__ unsigned int pk2bf(float lo, float hi) {
    return f2bf(lo) | (f2bf(hi) << 16);
}
__device__ __forceinline__ float bf2f(unsigned short h) {
    return __uint_as_float(((unsigned int)h) << 16);
}
__device__ __forceinline__ float sigf(float x) { return 1.f / (1.f + __expf(-x)); }

union FragBF { v16bf v; unsigned int u[8]; };

// ---------------------------------------------------------------------------
// Generic bf16-WMMA GEMM: C[M,N] = A[M,K] @ W[K,N], fp32 accumulate.
//  - A_BF16=true : raw bf16 A, moved by GLOBAL_LOAD_ASYNC_TO_LDS_B128 when
//    available, with K split at `ksplit` (k < ksplit -> A0, else A1) and a
//    row gather on A0 (physical row = m * a0_row_scale + a0_row_off, which
//    reproduces the reference's outs.reshape interleave: row = b2*128 + t2).
//  - A_BF16=false: fp32 A converted in the loader; this path is only used
//    split-free (ksplit == K) so it uses straight 16B vector loads and
//    packed b128 LDS stores.
//  - W is always fp32, converted and transposed into LDS. ksplit % 32 == 0,
//    so a 32-wide K tile never straddles the split.
// Block = 64x64 tile, 128 threads (4 waves), each wave a 2x2 of 16x16 frags,
// K staged 32-wide through double-buffered LDS (tile i+1 loads overlap the
// WMMAs of tile i). LDS rows padded to 40 ushorts (80 B): 16B-aligned for
// B128 async writes and bank-conflict-free for the fragment gathers
// (m*20 mod 64 distinct for m = 0..15).
// All call sites have M%64 == N%64 == K%32 == 0 (no bounds checks).
// ---------------------------------------------------------------------------
template<bool A_BF16, bool OUT_BF16>
__global__ __launch_bounds__(128) void gemm_wmma_kernel(
    const void* __restrict__ A0v, const void* __restrict__ A1v,
    const float* __restrict__ W0p, const float* __restrict__ W1p,
    void* __restrict__ Cv,
    int M, int N, int K, int ksplit,
    int lda0, int lda1, int ldw0, int ldw1, int ldc,
    int a0_row_scale, int a0_row_off)
{
    __shared__ unsigned short As[2][64][40];   // [buf][m][k], pad->80B rows
    __shared__ unsigned short Bs[2][64][40];   // [buf][n][k] (transposed)

    const int tid  = threadIdx.x;
    const int wave = tid >> 5;
    const int lane = tid & 31;
    const int hlf  = lane >> 4;   // 0: lanes 0-15, 1: lanes 16-31
    const int l16  = lane & 15;
    const int gm0  = blockIdx.y * 64;
    const int gn0  = blockIdx.x * 64;
    const int wm   = (wave >> 1) * 32;
    const int wn   = (wave & 1) * 32;

    const float*          A0f = (const float*)A0v;
    const unsigned short* A0h = (const unsigned short*)A0v;
    const unsigned short* A1h = (const unsigned short*)A1v;

    const v8f vzero = {0.f, 0.f, 0.f, 0.f, 0.f, 0.f, 0.f, 0.f};
    v8f acc[2][2];
    acc[0][0] = vzero; acc[0][1] = vzero; acc[1][0] = vzero; acc[1][1] = vzero;

    // ---- A tile loader: 64x32 bf16 into As[buf] ----
    auto loadA = [&](int k0, int buf) {
        if (A_BF16) {
#if HAS_ASYNC_LDS
            // Async DMA: 256 chunks of 16B (4 per row); 2 per thread.
            const bool lo = (k0 < ksplit);
#pragma unroll
            for (int c = 0; c < 2; ++c) {
                const int idx = tid + c * 128;    // 0..255
                const int m   = idx >> 2;
                const int q   = idx & 3;          // 16B quarter-row
                const int mg  = gm0 + m;
                const unsigned short* gp = lo
                    ? A0h + (size_t)(mg * a0_row_scale + a0_row_off) * (size_t)lda0
                          + (k0 + q * 8)
                    : A1h + (size_t)mg * (size_t)lda1 + ((k0 - ksplit) + q * 8);
                __builtin_amdgcn_global_load_async_to_lds_b128(
                    (async_v4i*)gp,
                    (lds_v4i_ptr)&As[buf][m][q * 8],
                    0, 0);
            }
#else
            const int base = tid * 16;
            const int m    = base >> 5;
            const int ks   = base & 31;
            const int mg   = gm0 + m;
            const bool lo  = (k0 < ksplit);
            const unsigned short* ap = lo
                ? A0h + (size_t)(mg * a0_row_scale + a0_row_off) * (size_t)lda0
                      + (k0 + ks)
                : A1h + (size_t)mg * (size_t)lda1 + ((k0 - ksplit) + ks);
            *(vu4*)&As[buf][m][ks]     = *(const vu4*)(ap + 0);
            *(vu4*)&As[buf][m][ks + 8] = *(const vu4*)(ap + 8);
#endif
        } else {
            // fp32 A, split-free (ksplit == K at every fp32 call site):
            // 4x16B vector loads, convert, 2x16B packed LDS stores.
            const int base = tid * 16;
            const int m    = base >> 5;
            const int ks   = base & 31;
            const int mg   = gm0 + m;
            const float* ap =
                A0f + (size_t)(mg * a0_row_scale + a0_row_off) * (size_t)lda0
                    + (k0 + ks);
            const vf4 f0 = *(const vf4*)(ap + 0);
            const vf4 f1 = *(const vf4*)(ap + 4);
            const vf4 f2 = *(const vf4*)(ap + 8);
            const vf4 f3 = *(const vf4*)(ap + 12);
            vu4 lo, hi;
            lo[0] = pk2bf(f0[0], f0[1]); lo[1] = pk2bf(f0[2], f0[3]);
            lo[2] = pk2bf(f1[0], f1[1]); lo[3] = pk2bf(f1[2], f1[3]);
            hi[0] = pk2bf(f2[0], f2[1]); hi[1] = pk2bf(f2[2], f2[3]);
            hi[2] = pk2bf(f3[0], f3[1]); hi[3] = pk2bf(f3[2], f3[3]);
            *(vu4*)&As[buf][m][ks]     = lo;
            *(vu4*)&As[buf][m][ks + 8] = hi;
        }
    };

    // ---- W tile loader: 32x64 fp32 -> bf16, transposed into Bs[buf][n][k].
    // Each thread owns a 2(k) x 8(n) sub-tile: 4x16B global loads, 8 packed
    // b32 LDS stores (k-pair per dword). The k-pair never straddles ksplit.
    auto loadW = [&](int k0, int buf) {
        const int kp = (tid & 15) * 2;    // k, k+1
        const int nb = (tid >> 4) * 8;    // 8 consecutive n
        const int gk = k0 + kp;
        const float* wr;
        int ldw;
        if (gk < ksplit) { wr = W0p + (size_t)gk * (size_t)ldw0 + gn0 + nb; ldw = ldw0; }
        else { wr = W1p + (size_t)(gk - ksplit) * (size_t)ldw1 + gn0 + nb; ldw = ldw1; }
        const vf4 r0a = *(const vf4*)(wr + 0);
        const vf4 r0b = *(const vf4*)(wr + 4);
        const vf4 r1a = *(const vf4*)(wr + ldw + 0);
        const vf4 r1b = *(const vf4*)(wr + ldw + 4);
#pragma unroll
        for (int i = 0; i < 4; ++i) {
            *(unsigned int*)&Bs[buf][nb + i][kp]     = pk2bf(r0a[i], r1a[i]);
            *(unsigned int*)&Bs[buf][nb + 4 + i][kp] = pk2bf(r0b[i], r1b[i]);
        }
    };

    // ---- wait for outstanding tile traffic, then workgroup barrier ----
    auto fence = [&]() {
#if HAS_ASYNC_LDS
        if (A_BF16) __builtin_amdgcn_s_wait_asynccnt(0);
#endif
        __syncthreads();
    };

    // ---- one 32-deep K step on buffer `buf` ----
    auto compute = [&](int buf) {
        FragBF af[2], bfr[2];
#pragma unroll
        for (int fm = 0; fm < 2; ++fm) {
            const int m = wm + fm * 16 + l16;
#pragma unroll
            for (int v = 0; v < 8; ++v) {
                // A 16x32 bf16: vgpr v<4 -> k=2v (+8*hlf), v>=4 -> k=16+2(v-4) (+8*hlf)
                const int kk = (v < 4 ? 2 * v : 16 + 2 * (v - 4)) + hlf * 8;
                af[fm].u[v] = *(const unsigned int*)&As[buf][m][kk];
            }
        }
#pragma unroll
        for (int fn = 0; fn < 2; ++fn) {
            const int n = wn + fn * 16 + l16;
#pragma unroll
            for (int v = 0; v < 8; ++v) {
                // B 32x16 bf16: vgpr v<4 -> k=2v (+8*hlf), v>=4 -> k=16+2(v-4) (+8*hlf)
                const int kk = (v < 4 ? 2 * v : 16 + 2 * (v - 4)) + hlf * 8;
                bfr[fn].u[v] = *(const unsigned int*)&Bs[buf][n][kk];
            }
        }
#pragma unroll
        for (int fm = 0; fm < 2; ++fm)
#pragma unroll
            for (int fn = 0; fn < 2; ++fn)
                acc[fm][fn] = __builtin_amdgcn_wmma_f32_16x16x32_bf16(
                    false, af[fm].v, false, bfr[fn].v,
                    (short)0, acc[fm][fn], false, false);
    };

    // ---- double-buffered main loop ----
    const int nk = K >> 5;
    loadA(0, 0);
    loadW(0, 0);
    fence();
    for (int i = 0; i < nk; ++i) {
        const int cur = i & 1;
        if (i + 1 < nk) {               // prefetch next tile into other buffer
            loadA((i + 1) << 5, cur ^ 1);
            loadW((i + 1) << 5, cur ^ 1);
        }
        compute(cur);
        fence();
    }

    // ---- store C (16x16 f32 D layout: vgpr r -> m = r + 8*hlf) ----
#pragma unroll
    for (int fm = 0; fm < 2; ++fm) {
#pragma unroll
        for (int fn = 0; fn < 2; ++fn) {
            const int n_l = gn0 + wn + fn * 16 + l16;
#pragma unroll
            for (int r = 0; r < 8; ++r) {
                const int m_l = gm0 + wm + fm * 16 + hlf * 8 + r;
                const float val = acc[fm][fn][r];
                if (OUT_BF16)
                    ((unsigned short*)Cv)[(size_t)m_l * ldc + n_l] =
                        (unsigned short)f2bf(val);
                else
                    ((float*)Cv)[(size_t)m_l * ldc + n_l] = val;
            }
        }
    }
}

// ---------------------------------------------------------------------------
// Zero persistent state at the start of every launch (harness does not).
// ---------------------------------------------------------------------------
__global__ void init_state_kernel(float* ctxf, float* cf, unsigned short* hprevb, int n)
{
    const int i = blockIdx.x * blockDim.x + threadIdx.x;
    if (i < n) { ctxf[i] = 0.f; cf[i] = 0.f; hprevb[i] = 0; }
}

// ---------------------------------------------------------------------------
// Step t: xin = [x_t, context] @ Wd + bd (Dense(1)), then LSTM cell 0 with
// zero initial state (h@U0 term vanishes): z = xin*W0 + b0,
// c0 = sig(i)*tanh(g), h0 = sig(o)*tanh(c0). One block per batch row.
// ---------------------------------------------------------------------------
__global__ __launch_bounds__(128) void xin_lstm0_kernel(
    const float* __restrict__ inputs, const float* __restrict__ Wd,
    const float* __restrict__ bd, const float* __restrict__ W0,
    const float* __restrict__ b0, const float* __restrict__ ctxf,
    unsigned short* __restrict__ h0b, int t)
{
    __shared__ float red[128];
    __shared__ float xin_s;
    const int b = blockIdx.x, tid = threadIdx.x;

    float s = 0.f;
    for (int j = tid; j < FF + UE; j += 128) {
        const float x = (j < FF) ? inputs[(b * TT + t) * FF + j]
                                 : ctxf[b * UE + (j - FF)];
        s += x * Wd[j];
    }
    red[tid] = s; __syncthreads();
    for (int off = 64; off > 0; off >>= 1) {
        if (tid < off) red[tid] += red[tid + off];
        __syncthreads();
    }
    if (tid == 0) xin_s = red[0] + bd[0];
    __syncthreads();
    const float xin = xin_s;

    for (int j = tid; j < DD; j += 128) {
        const float zi = xin * W0[j]          + b0[j];
        const float zg = xin * W0[2 * DD + j] + b0[2 * DD + j];
        const float zo = xin * W0[3 * DD + j] + b0[3 * DD + j];
        const float c0 = sigf(zi) * tanhf(zg);
        const float h0 = sigf(zo) * tanhf(c0);
        h0b[b * DD + j] = (unsigned short)f2bf(h0);
    }
}

// ---------------------------------------------------------------------------
// LSTM cell 1 gates (zero initial state): z1 = h0@W1 (from GEMM) + b1.
// Writes [h1,c1] bf16 for the Wq GEMM and h1 into outs[t*B+b][0:512].
// ---------------------------------------------------------------------------
__global__ void z1_gate_kernel(const float* __restrict__ z,
                               const float* __restrict__ b1,
                               unsigned short* __restrict__ h1c1,
                               unsigned short* __restrict__ outs, int t)
{
    const int idx = blockIdx.x * blockDim.x + threadIdx.x;
    if (idx >= BB * DD) return;
    const int b = idx >> 9, j = idx & 511;
    const float* zr = z + (size_t)b * (4 * DD);
    const float zi = zr[j]          + b1[j];
    const float zg = zr[2 * DD + j] + b1[2 * DD + j];
    const float zo = zr[3 * DD + j] + b1[3 * DD + j];
    const float c1 = sigf(zi) * tanhf(zg);
    const float h1 = sigf(zo) * tanhf(c1);
    h1c1[(size_t)b * 1024 + j]      = (unsigned short)f2bf(h1);
    h1c1[(size_t)b * 1024 + DD + j] = (unsigned short)f2bf(c1);
    outs[((size_t)t * BB + b) * 1024 + j] = (unsigned short)f2bf(h1);
}

// ---------------------------------------------------------------------------
// Additive attention for step t, one block per batch row b:
//   score[t'] = sum_a v[a] * tanh(q[b,a] + enc_proj[b,t',a])
//   w = softmax(score); ctx[u] = sum_t' w[t'] * enc[b,t',u]
// Writes ctx fp32 (next step's Dense input) and into outs[t*B+b][512:1024].
// ---------------------------------------------------------------------------
__global__ __launch_bounds__(128) void attention_kernel(
    const float* __restrict__ q, const float* __restrict__ v_att,
    const unsigned short* __restrict__ enc_proj, const float* __restrict__ enc,
    float* __restrict__ ctxf, unsigned short* __restrict__ outs, int t)
{
    __shared__ float qs[AA];
    __shared__ float vs[AA];
    __shared__ float red[128];
    __shared__ float wsm[TT];
    const int b = blockIdx.x, tid = threadIdx.x;

    for (int i = tid; i < AA; i += 128) { qs[i] = q[b * AA + i]; vs[i] = v_att[i]; }
    __syncthreads();

    // score for t' = tid (blockDim == TT == 128)
    const unsigned short* ep = enc_proj + ((size_t)b * TT + tid) * AA;
    float s = 0.f;
    for (int a = 0; a < AA; ++a)
        s += vs[a] * tanhf(qs[a] + bf2f(ep[a]));

    red[tid] = s; __syncthreads();
    for (int off = 64; off > 0; off >>= 1) {
        if (tid < off) red[tid] = fmaxf(red[tid], red[tid + off]);
        __syncthreads();
    }
    const float mx = red[0]; __syncthreads();
    const float e = __expf(s - mx);
    red[tid] = e; __syncthreads();
    for (int off = 64; off > 0; off >>= 1) {
        if (tid < off) red[tid] += red[tid + off];
        __syncthreads();
    }
    wsm[tid] = e / red[0];
    __syncthreads();

    for (int u = tid; u < UE; u += 128) {
        float c = 0.f;
        const float* er = enc + (size_t)b * TT * UE + u;
        for (int tp = 0; tp < TT; ++tp)
            c += wsm[tp] * er[(size_t)tp * UE];
        ctxf[b * UE + u] = c;
        outs[((size_t)t * BB + b) * 1024 + DD + u] = (unsigned short)f2bf(c);
    }
}

// ---------------------------------------------------------------------------
// Final stateful LSTM gates: z = [x_t,h]@[Wf;Uf] (from GEMM) + bf.
// c = sig(f)*c + sig(i)*tanh(g); h = sig(o)*tanh(c). Writes output row and
// bf16 h for the next step's GEMM.
// ---------------------------------------------------------------------------
__global__ void final_gate_kernel(const float* __restrict__ z,
                                  const float* __restrict__ bfv,
                                  float* __restrict__ cf,
                                  unsigned short* __restrict__ hprevb,
                                  float* __restrict__ out, int t)
{
    const int idx = blockIdx.x * blockDim.x + threadIdx.x;
    if (idx >= BB * DD) return;
    const int b = idx >> 9, j = idx & 511;
    const float* zr = z + (size_t)b * (4 * DD);
    const float zi = zr[j]          + bfv[j];
    const float zf = zr[DD + j]     + bfv[DD + j];
    const float zg = zr[2 * DD + j] + bfv[2 * DD + j];
    const float zo = zr[3 * DD + j] + bfv[3 * DD + j];
    const float c  = sigf(zf) * cf[idx] + sigf(zi) * tanhf(zg);
    const float h  = sigf(zo) * tanhf(c);
    cf[idx]     = c;
    hprevb[idx] = (unsigned short)f2bf(h);
    out[((size_t)b * TT + t) * DD + j] = h;   // [B,T,D]
}

// ---------------------------------------------------------------------------
extern "C" void kernel_launch(void* const* d_in, const int* in_sizes, int n_in,
                              void* d_out, int out_size, void* d_ws, size_t ws_size,
                              hipStream_t stream)
{
    (void)in_sizes; (void)n_in; (void)out_size; (void)ws_size;
    const float* inputs = (const float*)d_in[0];   // [256,128,32]
    const float* enc    = (const float*)d_in[1];   // [256,128,512]
    const float* Wd     = (const float*)d_in[2];   // [544]
    const float* bd     = (const float*)d_in[3];   // [1]
    const float* W0     = (const float*)d_in[4];   // [1,2048]
    const float* b0     = (const float*)d_in[6];   // [2048]   (U0 d_in[5] unused: h=0)
    const float* W1     = (const float*)d_in[7];   // [512,2048]
    const float* b1     = (const float*)d_in[9];   // [2048]   (U1 d_in[8] unused: h=0)
    const float* Wq     = (const float*)d_in[10];  // [1024,512]
    const float* Wk     = (const float*)d_in[11];  // [512,512]
    const float* v_att  = (const float*)d_in[12];  // [512]
    const float* Wf     = (const float*)d_in[13];  // [1024,2048]
    const float* Uf     = (const float*)d_in[14];  // [512,2048]
    const float* bfv    = (const float*)d_in[15];  // [2048]
    float* out = (float*)d_out;                    // [256,128,512]

    // workspace carve-up (~100 MB)
    size_t off = 0;
    char* base = (char*)d_ws;
    auto carve = [&](size_t bytes) {
        void* p = base + off;
        off += (bytes + 255) & ~(size_t)255;
        return p;
    };
    unsigned short* enc_proj = (unsigned short*)carve((size_t)BB * TT * AA * 2);   // 32 MB
    unsigned short* outs     = (unsigned short*)carve((size_t)BB * TT * 1024 * 2); // 64 MB
    float*          zbuf     = (float*)carve((size_t)BB * 4 * DD * 4);             // 2 MB
    float*          qbuf     = (float*)carve((size_t)BB * AA * 4);
    unsigned short* h0b      = (unsigned short*)carve((size_t)BB * DD * 2);
    unsigned short* h1c1     = (unsigned short*)carve((size_t)BB * 1024 * 2);
    float*          ctxf     = (float*)carve((size_t)BB * UE * 4);
    float*          cf       = (float*)carve((size_t)BB * DD * 4);
    unsigned short* hprevb   = (unsigned short*)carve((size_t)BB * DD * 2);

    // 0) zero carried state
    init_state_kernel<<<(BB * DD + 255) / 256, 256, 0, stream>>>(ctxf, cf, hprevb, BB * DD);

    // 1) enc_proj = encoder_output @ Wk  (state-independent, hoisted)
    gemm_wmma_kernel<false, true><<<dim3(AA / 64, (BB * TT) / 64), 128, 0, stream>>>(
        enc, nullptr, Wk, nullptr, enc_proj,
        BB * TT, AA, UE, /*ksplit*/ UE,
        /*lda0*/ UE, 0, /*ldw0*/ AA, 0, /*ldc*/ AA, 1, 0);

    // 2) sequential decoder scan
    for (int t = 0; t < TT; ++t) {
        xin_lstm0_kernel<<<BB, 128, 0, stream>>>(inputs, Wd, bd, W0, b0, ctxf, h0b, t);

        // z1 = h0 @ W1
        gemm_wmma_kernel<true, false><<<dim3((4 * DD) / 64, BB / 64), 128, 0, stream>>>(
            h0b, nullptr, W1, nullptr, zbuf,
            BB, 4 * DD, DD, DD, DD, 0, 4 * DD, 0, 4 * DD, 1, 0);

        z1_gate_kernel<<<(BB * DD + 255) / 256, 256, 0, stream>>>(zbuf, b1, h1c1, outs, t);

        // q = [h1,c1] @ Wq
        gemm_wmma_kernel<true, false><<<dim3(AA / 64, BB / 64), 128, 0, stream>>>(
            h1c1, nullptr, Wq, nullptr, qbuf,
            BB, AA, 2 * DD, 2 * DD, 2 * DD, 0, AA, 0, AA, 1, 0);

        attention_kernel<<<BB, 128, 0, stream>>>(qbuf, v_att, enc_proj, enc, ctxf, outs, t);
    }

    // 3) final stateful LSTM over the reshaped sequence:
    //    x row for output-row b2 at step t2 is outs row (b2*128 + t2).
    //    z = [x_t, h] @ [Wf; Uf] via K-split at 1024.
    for (int t = 0; t < TT; ++t) {
        gemm_wmma_kernel<true, false><<<dim3((4 * DD) / 64, BB / 64), 128, 0, stream>>>(
            outs, hprevb, Wf, Uf, zbuf,
            BB, 4 * DD, 1024 + DD, /*ksplit*/ 1024,
            /*lda0*/ 1024, /*lda1*/ DD, /*ldw0*/ 4 * DD, /*ldw1*/ 4 * DD, /*ldc*/ 4 * DD,
            /*a0_row_scale*/ TT, /*a0_row_off*/ t);

        final_gate_kernel<<<(BB * DD + 255) / 256, 256, 0, stream>>>(
            zbuf, bfv, cf, hprevb, out, t);
    }
}
